// STAttentionBlock_32547262169740
// MI455X (gfx1250) — compile-verified, matrix operand order checked by reading, not run on hardware
//
#include <hip/hip_runtime.h>
#include <math.h>

// ---------------------------------------------------------------------------
// Types for CDNA5 WMMA (wave32): v_wmma_f32_16x16x32_bf16
// ---------------------------------------------------------------------------
typedef __bf16 bf16;
typedef __attribute__((ext_vector_type(16))) __bf16 v16bf;
typedef __attribute__((ext_vector_type(8)))  __bf16 bf16x8;
typedef __attribute__((ext_vector_type(8)))  float  v8f;

// Pointer types for the async global->LDS builtin (param type from the
// compiler diagnostic: 'int __vector(4) *').  Address-space-qualified
// pointees so the casts produce real addrspacecasts; flat is a superset of
// AS1/AS3 so these also convert implicitly if the builtin takes generic ptrs.
typedef int v4i __attribute__((vector_size(16)));
typedef __attribute__((address_space(1))) v4i v4i_g;
typedef __attribute__((address_space(3))) v4i v4i_l;

static constexpr int NB   = 128;   // batch
static constexpr int CIN  = 64;
static constexpr int COUT = 128;
static constexpr int CI   = 32;    // C_INT
static constexpr int S    = 3;
static constexpr int T    = 64;
static constexpr int V    = 25;
static constexpr int MPOS = NB * T * V;          // 204800 positions
static constexpr size_t OUT_ELEMS = (size_t)NB * COUT * T * V;

#if defined(__has_builtin)
#if __has_builtin(__builtin_amdgcn_global_load_async_to_lds_b128)
#define HAVE_ASYNC_LDS 1
#endif
#endif

// ---------------------------------------------------------------------------
// Fragment loaders.
// ISA 16-bit A-matrix 16x32 layout (lane = M row; lanes 0-15: K {0..7,16..23},
// lanes 16-31: K {8..15,24..31}; 2 values per VGPR).  B fragments are loaded
// identically from B^T (N x K row-major), so conv weights [Cout][Cin] and our
// transposed-attention buffers feed directly.
// All call sites keep base + r*ld + 8*half 16-byte aligned.
// ---------------------------------------------------------------------------
static __device__ __forceinline__ v16bf frag_load(const bf16* __restrict__ base,
                                                  int ld, int lane) {
  const int half = lane >> 4;
  const int r    = lane & 15;
  const bf16* p = base + (size_t)r * ld + half * 8;
  bf16x8 lo = *(const bf16x8*)(p);        // K = off+0..7
  bf16x8 hi = *(const bf16x8*)(p + 16);   // K = off+16..23
  v16bf f;
#pragma unroll
  for (int i = 0; i < 8; ++i) { f[i] = lo[i]; f[i + 8] = hi[i]; }
  return f;
}

// Row-clamped variant: out-of-range rows duplicate a valid row; the
// corresponding D rows/cols are discarded by the epilogue.
static __device__ __forceinline__ v16bf frag_load_clamp(const bf16* __restrict__ base,
                                                        int ld, int lane, int maxr) {
  const int half = lane >> 4;
  int r = lane & 15;
  if (r > maxr) r = maxr;
  const bf16* p = base + (size_t)r * ld + half * 8;
  bf16x8 lo = *(const bf16x8*)(p);
  bf16x8 hi = *(const bf16x8*)(p + 16);
  v16bf f;
#pragma unroll
  for (int i = 0; i < 8; ++i) { f[i] = lo[i]; f[i + 8] = hi[i]; }
  return f;
}

static __device__ __forceinline__ v8f wmma_bf16(v16bf a, v16bf b, v8f c) {
  return __builtin_amdgcn_wmma_f32_16x16x32_bf16(false, a, false, b,
                                                 (short)0, c, false, false);
}

// ---------------------------------------------------------------------------
// Weight prep: f32 -> bf16 copy, and fused (bias, BN) -> per-channel scale/shift
// ---------------------------------------------------------------------------
__global__ void k_cast(const float* __restrict__ s, bf16* __restrict__ d, int n) {
  int i = blockIdx.x * blockDim.x + threadIdx.x;
  if (i < n) d[i] = (bf16)s[i];
}

__global__ void k_scaleshift(const float* __restrict__ bias,
                             const float* __restrict__ bn, // stacked [g,b,m,v] or null
                             float* __restrict__ ss, int C) {
  int c = blockIdx.x * blockDim.x + threadIdx.x;
  if (c >= C) return;
  float sc = 1.f, sh;
  if (bn) {
    float g = bn[c], be = bn[C + c], m = bn[2 * C + c], vv = bn[3 * C + c];
    sc = g * rsqrtf(vv + 1e-5f);
    sh = (bias[c] - m) * sc + be;
  } else {
    sh = bias[c];
  }
  ss[c] = sc;
  ss[C + c] = sh;
}

// ---------------------------------------------------------------------------
// x (f32 NCHW) -> bf16 in three layouts:
//   xnchw: [n][c][t][32]  (V padded to 32 with zeros; K=u contiguous for apply_s)
//   xnhwc: [n][t][v][64]  (K=c contiguous for the downsample convs)
//   xpe  : xnhwc + spatial positional encoding (qk_s input)
// ---------------------------------------------------------------------------
__global__ void k_convert_x(const float* __restrict__ x, bf16* __restrict__ xnchw,
                            bf16* __restrict__ xnhwc, bf16* __restrict__ xpe) {
  int i = blockIdx.x * blockDim.x + threadIdx.x;   // N*C*T*32 = 16777216
  if (i >= NB * CIN * T * 32) return;
  int v = i & 31;
  int t = (i >> 5) & 63;
  int c = (i >> 11) & 63;
  int n = i >> 17;
  float val = 0.f;
  if (v < V) val = x[(((size_t)n * CIN + c) * T + t) * V + v];
  xnchw[i] = (bf16)val;
  if (v < V) {
    size_t hw = (((size_t)n * T + t) * V + v) * CIN + c;
    xnhwc[hw] = (bf16)val;
    float ang = (float)v * __expf((float)(c & ~1) * -0.14391157f); // -ln(1e4)/64
    float pe = (c & 1) ? __cosf(ang) : __sinf(ang);
    xpe[hw] = (bf16)(val + pe);
  }
}

// ---------------------------------------------------------------------------
// Generic conv1x1 GEMM: out[M][N] = A[M][K] * W[N][K]^T, epilogue y*sc+sh.
// block = 256 (8 waves); wave -> 16x64 tile; grid = (M/128, N/64).
//  - 64xK weight strip staged once per block in LDS (async global->LDS when
//    available), B fragments then come from ds_load_b128.
//  - A fragments register-double-buffered so global loads overlap the WMMAs.
// ---------------------------------------------------------------------------
__global__ void __launch_bounds__(256) k_gemm(
    const bf16* __restrict__ A, const bf16* __restrict__ W,
    const float* __restrict__ ss, float* __restrict__ outF,
    bf16* __restrict__ outH, int K, int Ncols) {
  __shared__ bf16 sW[64 * 384];                 // up to 48 KB of the 320 KB WGP LDS
  const int tid = threadIdx.x;
  const int lane = tid & 31;
  const int wave = tid >> 5;
  const int row0 = blockIdx.x * 128 + wave * 16;
  const int col0 = blockIdx.y * 64;

  // Stage the 64-row weight strip for this block's column range.
  const bf16* wsrc = W + (size_t)col0 * K;
  const int chunks = (64 * K) / 8;              // 16-byte chunks
  for (int idx = tid; idx < chunks; idx += 256) {
#if defined(HAVE_ASYNC_LDS)
    __builtin_amdgcn_global_load_async_to_lds_b128((v4i_g*)(wsrc + idx * 8),
                                                   (v4i_l*)(sW + idx * 8), 0, 0);
#else
    *(bf16x8*)(sW + idx * 8) = *(const bf16x8*)(wsrc + idx * 8);
#endif
  }
#if defined(HAVE_ASYNC_LDS)
  asm volatile("s_wait_asynccnt 0" ::: "memory");
#endif
  __syncthreads();

  const bf16* Arow = A + (size_t)row0 * K;
  v8f acc[4] = {};
  v16bf a_cur = frag_load(Arow, K, lane);       // prologue: k = 0
  for (int k = 0; k < K; k += 32) {
    v16bf a_nxt = a_cur;
    if (k + 32 < K) {
      a_nxt = frag_load(Arow + k + 32, K, lane);          // in flight over WMMAs
      __builtin_prefetch(Arow + k + 64 + (lane & 15) * K, 0, 1);
    }
#pragma unroll
    for (int nt = 0; nt < 4; ++nt) {
      v16bf b = frag_load(sW + (size_t)(nt * 16) * K + k, K, lane);  // ds_load_b128
      acc[nt] = wmma_bf16(a_cur, b, acc[nt]);
    }
    a_cur = a_nxt;
  }

  const int cL = lane & 15;
  const int rH = (lane >> 4) * 8;
#pragma unroll
  for (int nt = 0; nt < 4; ++nt) {
    int c = col0 + nt * 16 + cL;
    float sc = ss[c], sh = ss[Ncols + c];
#pragma unroll
    for (int i = 0; i < 8; ++i) {
      int r = row0 + rH + i;
      float y = acc[nt][i] * sc + sh;
      size_t idx = (size_t)r * Ncols + c;
      if (outF) outF[idx] = y;
      if (outH) outH[idx] = (bf16)y;
    }
  }
}

// ---------------------------------------------------------------------------
// Spatial attention: per (n,s):  G[u][v] = sum_{t,c} q[c,t,u]*k[c,t,v]
// att = tanh(G/2048)*alpha + att0; stored TRANSPOSED bf16 [n][s][32v][32u],
// zero padded (so padded K lanes of apply_s multiply zeros).
// One wave per (n,s); 2x2 tiles of 16 cover 25x25.
// ---------------------------------------------------------------------------
__global__ void k_att_s(const bf16* __restrict__ qk, const float* __restrict__ alphas,
                        const float* __restrict__ att0, bf16* __restrict__ attT) {
  const int lane = threadIdx.x & 31;
  const int idx = blockIdx.x * (blockDim.x >> 5) + (threadIdx.x >> 5);
  const int n = idx / S, s = idx % S;
  v8f acc[4] = {};
  const size_t nb = (size_t)n * T * V * 192;
  for (int t = 0; t < T; ++t) {
    const bf16* qb = qk + nb + (size_t)t * V * 192 + s * CI;
    if (t + 1 < T)   // pull next t's rows toward the WGP while we compute
      __builtin_prefetch(qb + V * 192 + (size_t)(lane & 15) * 192, 0, 1);
    v16bf a0 = frag_load(qb, 192, lane);
    v16bf a1 = frag_load_clamp(qb + 16 * 192, 192, lane, 8);
    v16bf b0 = frag_load(qb + S * CI, 192, lane);
    v16bf b1 = frag_load_clamp(qb + 16 * 192 + S * CI, 192, lane, 8);
    acc[0] = wmma_bf16(a0, b0, acc[0]);
    acc[1] = wmma_bf16(a0, b1, acc[1]);
    acc[2] = wmma_bf16(a1, b0, acc[2]);
    acc[3] = wmma_bf16(a1, b1, acc[3]);
  }
  const float al = alphas[s];
  const int vL = lane & 15, uH = (lane >> 4) * 8;
#pragma unroll
  for (int iu = 0; iu < 2; ++iu)
#pragma unroll
    for (int iv = 0; iv < 2; ++iv) {
      v8f d = acc[iu * 2 + iv];
#pragma unroll
      for (int i = 0; i < 8; ++i) {
        int u = iu * 16 + uH + i;
        int v = iv * 16 + vL;
        float g = 0.f;
        if (u < V && v < V)
          g = tanhf(d[i] * (1.f / 2048.f)) * al + att0[(s * V + u) * V + v];
        attT[(((size_t)n * S + s) * 32 + v) * 32 + u] = (bf16)g;
      }
    }
}

// ---------------------------------------------------------------------------
// Spatial apply: per (n,t,s): D[c][v] = sum_u x[n,c,t,u] * att[u][v]
// A = xnchw (K=u contiguous, padded), B^T = attT rows v.  One wave per (n,t,s).
// Output bf16 NHWC with channel s*64+c (feeds w_outs GEMM, K=192).
// ---------------------------------------------------------------------------
__global__ void k_apply_s(const bf16* __restrict__ xp, const bf16* __restrict__ attT,
                          bf16* __restrict__ yap) {
  const int lane = threadIdx.x & 31;
  int idx = blockIdx.x * (blockDim.x >> 5) + (threadIdx.x >> 5);  // (n*T+t)*S+s
  int s = idx % S;
  int nt = idx / S;
  int t = nt & 63;
  int n = nt >> 6;
  const bf16* ab = attT + (((size_t)n * S + s) * 32) * 32;
  v16bf b0 = frag_load(ab, 32, lane);
  v16bf b1 = frag_load(ab + 16 * 32, 32, lane);
  const int vL = lane & 15, cH = (lane >> 4) * 8;
#pragma unroll
  for (int ct = 0; ct < 4; ++ct) {
    const bf16* abase = xp + ((size_t)(n * CIN + ct * 16) * T + t) * 32;
    v16bf a = frag_load(abase, T * 32, lane);
    v8f z = {};
    v8f d0 = wmma_bf16(a, b0, z);
    v8f d1 = wmma_bf16(a, b1, z);
#pragma unroll
    for (int i = 0; i < 8; ++i) {
      int c = ct * 16 + cH + i;
      size_t rowb = (size_t)(n * T + t) * V;
      if (vL < V)      yap[(rowb + vL) * 192 + s * CIN + c]        = (bf16)d0[i];
      if (16 + vL < V) yap[(rowb + 16 + vL) * 192 + s * CIN + c]   = (bf16)d1[i];
    }
  }
}

// ---------------------------------------------------------------------------
// Residual + LReLU elementwise over [MPOS][128]; optional bf16 NHWC copy,
// NHWC + temporal-PE copy (qk_t input), and [n][v][c][t] copy (apply_t input).
// ---------------------------------------------------------------------------
__global__ void k_add_lrelu(const float* __restrict__ d, const float* __restrict__ g,
                            bf16* __restrict__ o16, bf16* __restrict__ ope,
                            bf16* __restrict__ onvct) {
  int i = blockIdx.x * blockDim.x + threadIdx.x;   // MPOS*128
  if (i >= MPOS * COUT) return;
  float y = d[i] + g[i];
  y = y >= 0.f ? y : 0.1f * y;
  if (o16) o16[i] = (bf16)y;
  int c = i & 127;
  int pos = i >> 7;
  int t = (pos / V) & 63;
  if (ope) {
    float ang = (float)t * __expf((float)(c & ~1) * -0.071955784f); // -ln(1e4)/128
    float pe = (c & 1) ? __cosf(ang) : __sinf(ang);
    ope[i] = (bf16)(y + pe);
  }
  if (onvct) {
    int v = pos % V;
    int n = pos / (T * V);
    onvct[(((size_t)n * V + v) * COUT + c) * T + t] = (bf16)y;
  }
}

// ---------------------------------------------------------------------------
// Temporal attention: per (n,s): G[t][q] = sum_{c,v} q[c,t,v]*k[c,q,v]
// K-loop over v (25 steps of 32 channels, no padding needed).
// Block 128 = 4 waves; wave w owns t-tile w (16 rows) x full q (4 tiles).
// att stored transposed bf16 [n][s][q][t].
// ---------------------------------------------------------------------------
__global__ void k_att_t(const bf16* __restrict__ qk, const float* __restrict__ alphat,
                        const float* __restrict__ att0, bf16* __restrict__ attT) {
  const int lane = threadIdx.x & 31;
  const int wv = threadIdx.x >> 5;
  const int n = blockIdx.x / S, s = blockIdx.x % S;
  v8f acc[4] = {};
  const size_t nb = (size_t)n * T * V * 192;
  const int ldt = V * 192;   // stride between t rows = 4800 elems
  for (int v = 0; v < V; ++v) {
    const bf16* base = qk + nb + (size_t)v * 192 + s * CI;
    if (v + 1 < V)
      __builtin_prefetch(base + 192 + (size_t)(wv * 16 + (lane & 15)) * ldt, 0, 1);
    v16bf a = frag_load(base + (size_t)(wv * 16) * ldt, ldt, lane);
#pragma unroll
    for (int qt = 0; qt < 4; ++qt) {
      v16bf b = frag_load(base + S * CI + (size_t)(qt * 16) * ldt, ldt, lane);
      acc[qt] = wmma_bf16(a, b, acc[qt]);
    }
  }
  const float al = alphat[s];
  const int qL = lane & 15, tH = (lane >> 4) * 8;
#pragma unroll
  for (int qt = 0; qt < 4; ++qt)
#pragma unroll
    for (int i = 0; i < 8; ++i) {
      int t = wv * 16 + tH + i;
      int q = qt * 16 + qL;
      float g = tanhf(acc[qt][i] * (1.f / 800.f)) * al + att0[(s * T + t) * T + q];
      attT[(((size_t)n * S + s) * T + q) * T + t] = (bf16)g;
    }
}

// ---------------------------------------------------------------------------
// Temporal apply: per (n,s,v): D[q][c] = sum_t att[t][q] * y[n,c,t,v]
// A = attT_t rows q (K=t contiguous), B^T = y_nvct rows c (K=t contiguous).
// Block 128 = 4 waves; wave w owns q-tile w x 8 c-tiles.  Output channel
// s*128+c at position (n, q, v)  (feeds w_outt GEMM, K=384).
// ---------------------------------------------------------------------------
__global__ void k_apply_t(const bf16* __restrict__ attT, const bf16* __restrict__ ynvct,
                          bf16* __restrict__ zap) {
  const int lane = threadIdx.x & 31;
  const int wv = threadIdx.x >> 5;
  int b = blockIdx.x;              // (n*S+s)*V+v
  int v = b % V;
  int ns = b / V;
  int s = ns % S;
  int n = ns / S;
  v8f acc[8] = {};
  const bf16* Ab = attT + (((size_t)n * S + s) * T + wv * 16) * T;
  const bf16* Bb = ynvct + ((size_t)(n * V + v) * COUT) * T;
#pragma unroll
  for (int kk = 0; kk < T; kk += 32) {
    v16bf a = frag_load(Ab + kk, T, lane);
#pragma unroll
    for (int ct = 0; ct < 8; ++ct) {
      v16bf bb = frag_load(Bb + (size_t)(ct * 16) * T + kk, T, lane);
      acc[ct] = wmma_bf16(a, bb, acc[ct]);
    }
  }
  const int cL = lane & 15, qH = (lane >> 4) * 8;
#pragma unroll
  for (int ct = 0; ct < 8; ++ct)
#pragma unroll
    for (int i = 0; i < 8; ++i) {
      int q = wv * 16 + qH + i;
      int c = ct * 16 + cL;
      zap[((size_t)(n * T + q) * V + v) * 384 + s * COUT + c] = (bf16)acc[ct][i];
    }
}

// Final residual + LReLU, NHWC f32 -> NCHW f32 output.
__global__ void k_final(const float* __restrict__ d, const float* __restrict__ g,
                        float* __restrict__ out) {
  int i = blockIdx.x * blockDim.x + threadIdx.x;   // N*C*T*V
  if (i >= (int)OUT_ELEMS) return;
  int v = i % V;
  int t = (i / V) & 63;
  int c = (i / (T * V)) & 127;
  int n = i / (COUT * T * V);
  size_t src = ((size_t)(n * T + t) * V + v) * COUT + c;
  float y = d[src] + g[src];
  out[i] = y >= 0.f ? y : 0.1f * y;
}

// ---------------------------------------------------------------------------
// Host orchestration
// ---------------------------------------------------------------------------
extern "C" void kernel_launch(void* const* d_in, const int* in_sizes, int n_in,
                              void* d_out, int out_size, void* d_ws, size_t ws_size,
                              hipStream_t stream) {
  (void)in_sizes; (void)n_in; (void)out_size; (void)ws_size;
  const float* x      = (const float*)d_in[0];
  const float* w_qk_s = (const float*)d_in[1];
  const float* b_qk_s = (const float*)d_in[2];
  const float* alphas = (const float*)d_in[3];
  const float* att0s  = (const float*)d_in[4];
  const float* w_outs = (const float*)d_in[5];
  const float* b_outs = (const float*)d_in[6];
  const float* bn_outs= (const float*)d_in[7];
  const float* w_ffs  = (const float*)d_in[8];
  const float* b_ffs  = (const float*)d_in[9];
  const float* bn_ffs = (const float*)d_in[10];
  const float* w_ds1  = (const float*)d_in[11];
  const float* b_ds1  = (const float*)d_in[12];
  const float* bn_ds1 = (const float*)d_in[13];
  const float* w_ds2  = (const float*)d_in[14];
  const float* b_ds2  = (const float*)d_in[15];
  const float* bn_ds2 = (const float*)d_in[16];
  const float* w_qk_t = (const float*)d_in[17];
  const float* b_qk_t = (const float*)d_in[18];
  const float* alphat = (const float*)d_in[19];
  const float* att0t  = (const float*)d_in[20];
  const float* w_outt = (const float*)d_in[21];
  const float* b_outt = (const float*)d_in[22];
  const float* bn_outt= (const float*)d_in[23];
  const float* w_fft  = (const float*)d_in[24];
  const float* b_fft  = (const float*)d_in[25];
  const float* bn_fft = (const float*)d_in[26];
  const float* w_dt1  = (const float*)d_in[27];
  const float* b_dt1  = (const float*)d_in[28];
  const float* bn_dt1 = (const float*)d_in[29];
  const float* w_dt2  = (const float*)d_in[30];
  const float* b_dt2  = (const float*)d_in[31];
  const float* bn_dt2 = (const float*)d_in[32];

  char* base = (char*)d_ws;
  size_t off = 0;
  auto take = [&](size_t bytes) -> char* {
    char* p = base + off;
    off = (off + bytes + 255) & ~(size_t)255;
    return p;
  };

  // bf16 weights (N x K row-major, as the B-fragment loader expects)
  bf16* wqs16   = (bf16*)take(192 * 64 * 2);
  bf16* wouts16 = (bf16*)take(128 * 192 * 2);
  bf16* wffs16  = (bf16*)take(128 * 128 * 2);
  bf16* wds1_16 = (bf16*)take(128 * 64 * 2);
  bf16* wds2_16 = (bf16*)take(128 * 64 * 2);
  bf16* wqkt16  = (bf16*)take(192 * 128 * 2);
  bf16* woutt16 = (bf16*)take(128 * 384 * 2);
  bf16* wfft16  = (bf16*)take(128 * 128 * 2);
  bf16* wdt1_16 = (bf16*)take(128 * 128 * 2);
  bf16* wdt2_16 = (bf16*)take(128 * 128 * 2);
  // fused scale/shift per conv
  float* ss_qks  = (float*)take(2 * 192 * 4);
  float* ss_outs = (float*)take(2 * 128 * 4);
  float* ss_ds1  = (float*)take(2 * 128 * 4);
  float* ss_ffs  = (float*)take(2 * 128 * 4);
  float* ss_ds2  = (float*)take(2 * 128 * 4);
  float* ss_qkt  = (float*)take(2 * 192 * 4);
  float* ss_outt = (float*)take(2 * 128 * 4);
  float* ss_dt1  = (float*)take(2 * 128 * 4);
  float* ss_fft  = (float*)take(2 * 128 * 4);
  float* ss_dt2  = (float*)take(2 * 128 * 4);
  // activations
  bf16* xnchw = (bf16*)take((size_t)NB * CIN * T * 32 * 2);     // V padded to 32
  bf16* xnhwc = (bf16*)take((size_t)MPOS * CIN * 2);
  bf16* xpe   = (bf16*)take((size_t)MPOS * CIN * 2);
  bf16* qbuf  = (bf16*)take((size_t)MPOS * 192 * 2);            // qk_s -> yap -> qk_t
  bf16* attTs = (bf16*)take((size_t)NB * S * 32 * 32 * 2);
  bf16* attTt = (bf16*)take((size_t)NB * S * T * T * 2);
  float* F1   = (float*)take((size_t)MPOS * COUT * 4);
  float* F2   = (float*)take((size_t)MPOS * COUT * 4);
  bf16* y16   = (bf16*)take((size_t)MPOS * COUT * 2);
  bf16* ype   = (bf16*)take((size_t)MPOS * COUT * 2);           // reused as z16
  bf16* ynvct = (bf16*)take((size_t)NB * V * COUT * T * 2);
  bf16* zap   = (bf16*)take((size_t)MPOS * 384 * 2);
  bf16* z16   = ype;

  auto cast = [&](const float* s, bf16* d, int n) {
    k_cast<<<(n + 255) / 256, 256, 0, stream>>>(s, d, n);
  };
  cast(w_qk_s, wqs16, 192 * 64);
  cast(w_outs, wouts16, 128 * 192);
  cast(w_ffs, wffs16, 128 * 128);
  cast(w_ds1, wds1_16, 128 * 64);
  cast(w_ds2, wds2_16, 128 * 64);
  cast(w_qk_t, wqkt16, 192 * 128);
  cast(w_outt, woutt16, 128 * 384);
  cast(w_fft, wfft16, 128 * 128);
  cast(w_dt1, wdt1_16, 128 * 128);
  cast(w_dt2, wdt2_16, 128 * 128);

  auto ssk = [&](const float* b, const float* bn, float* ss, int C) {
    k_scaleshift<<<(C + 63) / 64, 64, 0, stream>>>(b, bn, ss, C);
  };
  ssk(b_qk_s, nullptr, ss_qks, 192);
  ssk(b_outs, bn_outs, ss_outs, 128);
  ssk(b_ds1, bn_ds1, ss_ds1, 128);
  ssk(b_ffs, bn_ffs, ss_ffs, 128);
  ssk(b_ds2, bn_ds2, ss_ds2, 128);
  ssk(b_qk_t, nullptr, ss_qkt, 192);
  ssk(b_outt, bn_outt, ss_outt, 128);
  ssk(b_dt1, bn_dt1, ss_dt1, 128);
  ssk(b_fft, bn_fft, ss_fft, 128);
  ssk(b_dt2, bn_dt2, ss_dt2, 128);

  k_convert_x<<<(NB * CIN * T * 32) / 256, 256, 0, stream>>>(x, xnchw, xnhwc, xpe);

  const int gx = MPOS / 128;   // 1600
  // ---- spatial block ----
  k_gemm<<<dim3(gx, 3), 256, 0, stream>>>(xpe, wqs16, ss_qks, nullptr, qbuf, 64, 192);
  k_att_s<<<(NB * S) / 4, 128, 0, stream>>>(qbuf, alphas, att0s, attTs);
  k_apply_s<<<(NB * T * S) / 4, 128, 0, stream>>>(xnchw, attTs, qbuf);
  k_gemm<<<dim3(gx, 2), 256, 0, stream>>>(qbuf, wouts16, ss_outs, F1, nullptr, 192, 128);
  k_gemm<<<dim3(gx, 2), 256, 0, stream>>>(xnhwc, wds1_16, ss_ds1, F2, nullptr, 64, 128);
  k_add_lrelu<<<(MPOS * COUT) / 256, 256, 0, stream>>>(F2, F1, y16, nullptr, nullptr);
  k_gemm<<<dim3(gx, 2), 256, 0, stream>>>(y16, wffs16, ss_ffs, F1, nullptr, 128, 128);
  k_gemm<<<dim3(gx, 2), 256, 0, stream>>>(xnhwc, wds2_16, ss_ds2, F2, nullptr, 64, 128);
  k_add_lrelu<<<(MPOS * COUT) / 256, 256, 0, stream>>>(F2, F1, y16, ype, ynvct);
  // ---- temporal block ----
  k_gemm<<<dim3(gx, 3), 256, 0, stream>>>(ype, wqkt16, ss_qkt, nullptr, qbuf, 128, 192);
  k_att_t<<<NB * S, 128, 0, stream>>>(qbuf, alphat, att0t, attTt);
  k_apply_t<<<NB * S * V, 128, 0, stream>>>(attTt, ynvct, zap);
  k_gemm<<<dim3(gx, 2), 256, 0, stream>>>(zap, woutt16, ss_outt, F1, nullptr, 384, 128);
  k_gemm<<<dim3(gx, 2), 256, 0, stream>>>(y16, wdt1_16, ss_dt1, F2, nullptr, 128, 128);
  k_add_lrelu<<<(MPOS * COUT) / 256, 256, 0, stream>>>(F2, F1, z16, nullptr, nullptr);
  k_gemm<<<dim3(gx, 2), 256, 0, stream>>>(z16, wfft16, ss_fft, F1, nullptr, 128, 128);
  k_gemm<<<dim3(gx, 2), 256, 0, stream>>>(y16, wdt2_16, ss_dt2, F2, nullptr, 128, 128);
  k_final<<<(int)(OUT_ELEMS / 256), 256, 0, stream>>>(F2, F1, (float*)d_out);
}
